// GCN_29540785062187
// MI455X (gfx1250) — matrix operand order repaired and checked by reading, not compiled, not run on hardware
//
#include <hip/hip_runtime.h>

#define NNODES 100000
#define FIN    512
#define HID    16
#define NCLS   40
#define NEDGE  3200000

typedef __attribute__((ext_vector_type(16))) __bf16 v16bf;
typedef __attribute__((ext_vector_type(8)))  float  v8f;

// ---------------------------------------------------------------------------
// 0) init: deg = 1.0 (self loop), agg = 0
// ---------------------------------------------------------------------------
__global__ void gcn_init(float* __restrict__ deg, float* __restrict__ agg) {
    int i = blockIdx.x * blockDim.x + threadIdx.x;
    if (i < NNODES * HID) agg[i] = 0.0f;
    if (i < NNODES)       deg[i] = 1.0f;
}

// ---------------------------------------------------------------------------
// 1) degree accumulation over dst
// ---------------------------------------------------------------------------
__global__ void gcn_deg(const long long* __restrict__ dst, float* __restrict__ deg) {
    int e = blockIdx.x * blockDim.x + threadIdx.x;
    if (e < NEDGE) atomicAdd(&deg[(int)dst[e]], 1.0f);
}

// ---------------------------------------------------------------------------
// 2) dinv = rsqrt(deg)   (deg >= 1 always due to self loop)
// ---------------------------------------------------------------------------
__global__ void gcn_dinv(float* __restrict__ deg) {
    int i = blockIdx.x * blockDim.x + threadIdx.x;
    if (i < NNODES) deg[i] = rsqrtf(deg[i]);
}

// ---------------------------------------------------------------------------
// 3) H = x @ W1 via v_wmma_f32_16x16x32_bf16. One wave = one 16-node tile,
//    full K=512 in 16 WMMA steps. W1 pre-transposed to bf16 [n][k] in LDS.
// ---------------------------------------------------------------------------
__global__ void __launch_bounds__(256) gcn_gemm1(const float* __restrict__ x,
                                                 const float* __restrict__ W1,
                                                 float* __restrict__ H) {
    __shared__ __attribute__((aligned(32))) __bf16 w[HID][FIN];  // 16 KB, [n][k]
    for (int i = threadIdx.x; i < FIN * HID; i += 256) {
        int k = i >> 4;          // W1 is [k][n] row-major
        int n = i & 15;
        w[n][k] = (__bf16)W1[i];
    }
    __syncthreads();

    const int wave = threadIdx.x >> 5;
    const int lane = threadIdx.x & 31;
    const int tile = blockIdx.x * 8 + wave;
    if (tile >= NNODES / 16) return;

    const int m  = lane & 15;     // row within tile (A), column N (B/C/D)
    const int hs = lane >> 4;     // lane half select

    const float* __restrict__ xr = x + (size_t)(tile * 16 + m) * FIN;
    v8f acc = {};

    for (int k0 = 0; k0 < FIN; k0 += 32) {
        // ---- A tile: lane<16 -> K {k0..k0+15}; lane>=16 -> K {k0+16..k0+31}
        const int ka = k0 + hs * 16;
        float4 a0 = *(const float4*)(xr + ka);
        float4 a1 = *(const float4*)(xr + ka + 4);
        float4 a2 = *(const float4*)(xr + ka + 8);
        float4 a3 = *(const float4*)(xr + ka + 12);
        v16bf A;
        A[0]  = (__bf16)a0.x; A[1]  = (__bf16)a0.y; A[2]  = (__bf16)a0.z; A[3]  = (__bf16)a0.w;
        A[4]  = (__bf16)a1.x; A[5]  = (__bf16)a1.y; A[6]  = (__bf16)a1.z; A[7]  = (__bf16)a1.w;
        A[8]  = (__bf16)a2.x; A[9]  = (__bf16)a2.y; A[10] = (__bf16)a2.z; A[11] = (__bf16)a2.w;
        A[12] = (__bf16)a3.x; A[13] = (__bf16)a3.y; A[14] = (__bf16)a3.z; A[15] = (__bf16)a3.w;

        // ---- B tile: lane holds column n=m, K = k0 + hs*16 + {0..15} (contiguous 32B in LDS)
        v16bf B = *(const v16bf*)&w[m][k0 + hs * 16];

        acc = __builtin_amdgcn_wmma_f32_16x16x32_bf16(
            /*neg_a=*/false, A, /*neg_b=*/false, B,
            /*c_mod=*/(short)0, acc, /*reuse_a=*/false, /*reuse_b=*/false);
    }

    // C/D layout: VGPR r -> row (r + 8*hs), col m
    float* __restrict__ hrow = H + (size_t)tile * 16 * HID;
#pragma unroll
    for (int r = 0; r < 8; ++r)
        hrow[(r + hs * 8) * HID + m] = acc[r];
}

// ---------------------------------------------------------------------------
// 4) edge scatter: 16 lanes per edge (one per feature), coalesced 64B
//    gather + 64B of contiguous f32 atomics. Used for both layers.
// ---------------------------------------------------------------------------
__global__ void gcn_scatter(const long long* __restrict__ src,
                            const long long* __restrict__ dst,
                            const float* __restrict__ dinv,
                            const float* __restrict__ h,
                            float* __restrict__ agg) {
    int t = blockIdx.x * blockDim.x + threadIdx.x;   // E*16 = 51.2M < 2^31
    int e = t >> 4;
    int f = t & 15;
    if (e < NEDGE) {
        int s = (int)src[e];
        int d = (int)dst[e];
        float nm = dinv[s] * dinv[d];
        atomicAdd(&agg[(size_t)d * HID + f], h[(size_t)s * HID + f] * nm);
    }
}

// ---------------------------------------------------------------------------
// 5) finalize layer 1: add self-loop + bias, ReLU; re-zero agg for layer 2
// ---------------------------------------------------------------------------
__global__ void gcn_fin1(const float* __restrict__ H, const float* __restrict__ dinv,
                         const float* __restrict__ b1, float* __restrict__ agg,
                         float* __restrict__ h1) {
    int i = blockIdx.x * blockDim.x + threadIdx.x;
    if (i >= NNODES * HID) return;
    int n = i >> 4;
    int f = i & 15;
    float di = dinv[n];
    float v  = agg[i] + H[i] * di * di + b1[f];
    h1[i]  = v > 0.0f ? v : 0.0f;
    agg[i] = 0.0f;                 // ready for layer-2 scatter
}

// ---------------------------------------------------------------------------
// 6) finalize layer 2: self-loop, 16x40 transform (+b2), log_softmax.
//    (aggregation commutes with the linear W2 transform)
// ---------------------------------------------------------------------------
__global__ void __launch_bounds__(256) gcn_fin2(const float* __restrict__ h1,
                                                const float* __restrict__ agg,
                                                const float* __restrict__ dinv,
                                                const float* __restrict__ W2,
                                                const float* __restrict__ b2,
                                                float* __restrict__ out) {
    __shared__ float w2[HID * NCLS];
    __shared__ float bb[NCLS];
    for (int i = threadIdx.x; i < HID * NCLS; i += blockDim.x) w2[i] = W2[i];
    for (int i = threadIdx.x; i < NCLS; i += blockDim.x)       bb[i] = b2[i];
    __syncthreads();

    int n = blockIdx.x * blockDim.x + threadIdx.x;
    if (n >= NNODES) return;

    float di = dinv[n];
    float d2 = di * di;

    float v[HID];
#pragma unroll
    for (int f = 0; f < HID; ++f)
        v[f] = agg[(size_t)n * HID + f] + h1[(size_t)n * HID + f] * d2;

    float z[NCLS];
#pragma unroll
    for (int c = 0; c < NCLS; ++c) z[c] = bb[c];
#pragma unroll
    for (int f = 0; f < HID; ++f) {
        float vf = v[f];
#pragma unroll
        for (int c = 0; c < NCLS; ++c) z[c] = fmaf(vf, w2[f * NCLS + c], z[c]);
    }

    float mx = z[0];
#pragma unroll
    for (int c = 1; c < NCLS; ++c) mx = fmaxf(mx, z[c]);
    float s = 0.0f;
#pragma unroll
    for (int c = 0; c < NCLS; ++c) s += __expf(z[c] - mx);
    float lse = mx + __logf(s);

    float* o = out + (size_t)n * NCLS;
#pragma unroll
    for (int c = 0; c < NCLS; ++c) o[c] = z[c] - lse;
}

// ---------------------------------------------------------------------------
// launcher
// ---------------------------------------------------------------------------
extern "C" void kernel_launch(void* const* d_in, const int* in_sizes, int n_in,
                              void* d_out, int out_size, void* d_ws, size_t ws_size,
                              hipStream_t stream) {
    const float*     x   = (const float*)d_in[0];
    const long long* ei  = (const long long*)d_in[1];   // int64 edge_index [2, E]
    const float*     W1  = (const float*)d_in[2];
    const float*     b1  = (const float*)d_in[3];
    const float*     W2  = (const float*)d_in[4];
    const float*     b2  = (const float*)d_in[5];
    const long long* src = ei;
    const long long* dst = ei + NEDGE;
    float*           out = (float*)d_out;

    // workspace layout (256B-aligned, ~19.6 MB total)
    char*  ws   = (char*)d_ws;
    float* dinv = (float*)(ws);              // N floats (deg -> dinv in place)
    float* H    = (float*)(ws +   400128);   // N*16
    float* agg  = (float*)(ws +  6800384);   // N*16
    float* h1   = (float*)(ws + 13200640);   // N*16

    const int nf    = NNODES * HID;                       // 1.6M
    const int sgrd  = (NEDGE * HID) / 256;                // 200000 blocks

    gcn_init   <<<(nf + 255) / 256,       256, 0, stream>>>(dinv, agg);
    gcn_deg    <<<(NEDGE + 255) / 256,    256, 0, stream>>>(dst, dinv);
    gcn_dinv   <<<(NNODES + 255) / 256,   256, 0, stream>>>(dinv);
    gcn_gemm1  <<<(NNODES / 16 + 7) / 8,  256, 0, stream>>>(x, W1, H);
    gcn_scatter<<<sgrd,                   256, 0, stream>>>(src, dst, dinv, H, agg);
    gcn_fin1   <<<(nf + 255) / 256,       256, 0, stream>>>(H, dinv, b1, agg, h1);
    gcn_scatter<<<sgrd,                   256, 0, stream>>>(src, dst, dinv, h1, agg);
    gcn_fin2   <<<(NNODES + 255) / 256,   256, 0, stream>>>(h1, agg, dinv, W2, b2, out);
}
